// GraphMathSolver_42099269435540
// MI455X (gfx1250) — matrix operand order repaired and checked by reading, not compiled, not run on hardware
//
#include <hip/hip_runtime.h>
#include <hip/hip_bf16.h>

// ---------------------------------------------------------------------------
// CDNA5 (gfx1250) GNN forward.
//   - Heavy GEMMs on v_wmma_f32_16x16x32_bf16 (bf16 in, f32 accum).
//   - msg kernel: 32 edges / workgroup, B-fragments reused across 2 M-tiles.
//   - A-panel gathers use GLOBAL_LOAD_ASYNC_TO_LDS_B128 (ASYNCcnt) when the
//     builtin is available; synchronous b128 copy otherwise.
// ---------------------------------------------------------------------------

typedef __attribute__((ext_vector_type(16))) __bf16 v16bf;
typedef __attribute__((ext_vector_type(8)))  __bf16 v8bf;
typedef __attribute__((ext_vector_type(8)))  float  v8f;

__device__ __forceinline__ v8f wmma_bf16(v16bf a, v16bf b, v8f c) {
  return __builtin_amdgcn_wmma_f32_16x16x32_bf16(false, a, false, b, (short)0, c,
                                                 false, false);
}

// A-matrix 16x32 bf16 fragment from a row-major LDS panel [16][ld].
__device__ __forceinline__ v16bf ldsA_frag(const __bf16* panel, int ld, int kt, int lane) {
  const int row = lane & 15;
  const int grp = lane >> 4;
  const int k1  = kt * 32 + grp * 8;
  const v8bf lo = *(const v8bf*)(panel + row * ld + k1);
  const v8bf hi = *(const v8bf*)(panel + row * ld + k1 + 16);
  return __builtin_shufflevector(lo, hi, 0, 1, 2, 3, 4, 5, 6, 7,
                                 8, 9, 10, 11, 12, 13, 14, 15);
}

// B-matrix 32x16 bf16 fragment, fragment-ordered: [kt][nt][lane][16 bf16].
__device__ __forceinline__ v16bf b_frag(const __bf16* wf, int kt, int nt, int lane) {
  return *(const v16bf*)(wf + (((size_t)kt * 16 + nt) * 32 + lane) * 16);
}

// 16-byte global -> LDS copy.  Async (ASYNCcnt-tracked) when available.
// Builtin signature (from hipcc diagnostic): first param is int32x4* in AS(1)
// (printed "__device__"), second is the LDS destination.
__device__ __forceinline__ void async_copy_b128(const void* gsrc, void* ldst) {
#if __has_builtin(__builtin_amdgcn_global_load_async_to_lds_b128)
  typedef int i32x4 __attribute__((vector_size(16)));
  __builtin_amdgcn_global_load_async_to_lds_b128(
      (__attribute__((address_space(1))) i32x4*)gsrc,
      (__attribute__((address_space(3))) i32x4*)ldst,
      0, 0);
#else
  *(uint4*)ldst = *(const uint4*)gsrc;
#endif
}

__device__ __forceinline__ void wait_async0() {
#if __has_builtin(__builtin_amdgcn_s_wait_asynccnt)
  __builtin_amdgcn_s_wait_asynccnt(0);
#elif defined(__gfx1250__)
  asm volatile("s_wait_asynccnt 0" ::: "memory");
#endif
}

// ---------------------------------------------------------------------------
// Weight conversion to B-fragment-ordered bf16.
// ---------------------------------------------------------------------------
__global__ __launch_bounds__(256) void conv_wfrag(const float* __restrict__ W,
                                                  __bf16* __restrict__ out,
                                                  int K, int Nout) {
  int gid = blockIdx.x * 256 + threadIdx.x;
  if (gid >= K * Nout) return;
  int p    = gid & 15;
  int lane = (gid >> 4) & 31;
  int tile = gid >> 9;
  int NT   = Nout >> 4;
  int nt   = tile % NT;
  int kt   = tile / NT;
  int k    = kt * 32 + (lane >> 4) * 8 + (p & 7) + (p >> 3) * 16;
  int n    = nt * 16 + (lane & 15);
  out[gid] = (__bf16)W[(size_t)k * Nout + n];
}

__global__ __launch_bounds__(256) void conv_bf16(const float* __restrict__ src,
                                                 __bf16* __restrict__ dst, int n) {
  int i = blockIdx.x * 256 + threadIdx.x;
  if (i < n) dst[i] = (__bf16)src[i];
}

__global__ __launch_bounds__(256) void zero_f32(float* __restrict__ p, int n) {
  int i = blockIdx.x * 256 + threadIdx.x;
  if (i < n) p[i] = 0.f;
}

// ---------------------------------------------------------------------------
// x = node_features @ node_W + node_b (16 rows / WG, K=128); writes f32 + bf16.
// ---------------------------------------------------------------------------
__global__ __launch_bounds__(256) void node_encode(const float* __restrict__ nf,
                                                   const __bf16* __restrict__ Wf,
                                                   const float* __restrict__ bias,
                                                   float* __restrict__ x,
                                                   __bf16* __restrict__ xb) {
  __shared__ __bf16 A[16 * 128];
  const int tid = threadIdx.x, lane = tid & 31, wave = tid >> 5;
  const int r0 = blockIdx.x * 16;
  for (int j = 0; j < 8; ++j) {
    int flat = tid + j * 256;
    int r = flat >> 7, c = flat & 127;
    A[flat] = (__bf16)nf[(size_t)(r0 + r) * 128 + c];
  }
  __syncthreads();
  for (int t = 0; t < 2; ++t) {
    int nt = wave * 2 + t;
    v8f acc = {};
    for (int kt = 0; kt < 4; ++kt)
      acc = wmma_bf16(ldsA_frag(A, 128, kt, lane), b_frag(Wf, kt, nt, lane), acc);
    int col = nt * 16 + (lane & 15);
    int rb  = (lane >> 4) * 8;
    float bv = bias[col];
    for (int i = 0; i < 8; ++i) {
      size_t idx = (size_t)(r0 + rb + i) * 256 + col;
      float v = acc[i] + bv;
      x[idx]  = v;
      xb[idx] = (__bf16)v;
    }
  }
}

// ---------------------------------------------------------------------------
// Per-edge message MLP: 32 edges / workgroup (two 16-row M-tiles).
//   A-panel (LDS bf16 32x768) = [x[src] | x[dst] | edge_attr] via async copies.
//   B-fragments loaded once per (kt,nt) and reused for both M-tiles.
// ---------------------------------------------------------------------------
__global__ __launch_bounds__(256) void msg_kernel(
    const __bf16* __restrict__ xb, const long long* __restrict__ ei,
    const __bf16* __restrict__ efb, const __bf16* __restrict__ edgeWf,
    const float* __restrict__ edge_b, const __bf16* __restrict__ msgW1f,
    const float* __restrict__ msg_b1, const __bf16* __restrict__ msgW2f,
    const float* __restrict__ msg_b2, float* __restrict__ agg, int layer, int E) {
  __shared__ __bf16 A[32 * 768];    // 48 KB
  __shared__ __bf16 Hd[32 * 256];   // 16 KB
  __shared__ __bf16 EF[32 * 64];    //  4 KB
  __shared__ int sSrc[32], sDst[32];
  const int tid = threadIdx.x, lane = tid & 31, wave = tid >> 5;
  const size_t e0 = (size_t)blockIdx.x * 32;
  if (tid < 32) {
    sSrc[tid] = (int)ei[e0 + tid];
    sDst[tid] = (int)ei[(size_t)E + e0 + tid];
  }
  __syncthreads();
  // async gathers: each wave copies 4 src rows + 4 dst rows (512 B each)
  for (int j = 0; j < 4; ++j) {
    int r = wave * 4 + j;
    async_copy_b128(xb + (size_t)sSrc[r] * 256 + lane * 8,
                    &A[r * 768 + lane * 8]);
    async_copy_b128(xb + (size_t)sDst[r] * 256 + lane * 8,
                    &A[r * 768 + 256 + lane * 8]);
  }
  // edge features: 32x64 bf16 = 4 KB, one 512 B chunk per wave
  async_copy_b128(efb + e0 * 64 + wave * 256 + lane * 8,
                  &EF[wave * 256 + lane * 8]);
  wait_async0();
  __syncthreads();
  // edge encoder -> A[:, 512:768]  (K = 64)
  for (int t = 0; t < 2; ++t) {
    int nt = wave * 2 + t;
    v8f a0 = {}, a1 = {};
    for (int kt = 0; kt < 2; ++kt) {
      v16bf b = b_frag(edgeWf, kt, nt, lane);
      a0 = wmma_bf16(ldsA_frag(EF, 64, kt, lane), b, a0);
      a1 = wmma_bf16(ldsA_frag(EF + 16 * 64, 64, kt, lane), b, a1);
    }
    int col = nt * 16 + (lane & 15), rb = (lane >> 4) * 8;
    float bv = edge_b[col];
    for (int i = 0; i < 8; ++i) {
      A[(rb + i) * 768 + 512 + col]        = (__bf16)(a0[i] + bv);
      A[(16 + rb + i) * 768 + 512 + col]   = (__bf16)(a1[i] + bv);
    }
  }
  __syncthreads();
  // GEMM1: hidden = relu(A @ W1 + b1), K = 768, B reused across 2 M-tiles
  const __bf16* W1 = msgW1f + (size_t)layer * (24 * 16 * 512);
  for (int t = 0; t < 2; ++t) {
    int nt = wave * 2 + t;
    v8f a0 = {}, a1 = {};
    for (int kt = 0; kt < 24; ++kt) {
      v16bf b = b_frag(W1, kt, nt, lane);
      a0 = wmma_bf16(ldsA_frag(A, 768, kt, lane), b, a0);
      a1 = wmma_bf16(ldsA_frag(A + 16 * 768, 768, kt, lane), b, a1);
    }
    int col = nt * 16 + (lane & 15), rb = (lane >> 4) * 8;
    float bv = msg_b1[layer * 256 + col];
    for (int i = 0; i < 8; ++i) {
      Hd[(rb + i) * 256 + col]      = (__bf16)fmaxf(a0[i] + bv, 0.f);
      Hd[(16 + rb + i) * 256 + col] = (__bf16)fmaxf(a1[i] + bv, 0.f);
    }
  }
  __syncthreads();
  // GEMM2: msg = hidden @ W2 + b2, K = 256; scatter-add to agg[dst]
  const __bf16* W2 = msgW2f + (size_t)layer * (8 * 16 * 512);
  for (int t = 0; t < 2; ++t) {
    int nt = wave * 2 + t;
    v8f a0 = {}, a1 = {};
    for (int kt = 0; kt < 8; ++kt) {
      v16bf b = b_frag(W2, kt, nt, lane);
      a0 = wmma_bf16(ldsA_frag(Hd, 256, kt, lane), b, a0);
      a1 = wmma_bf16(ldsA_frag(Hd + 16 * 256, 256, kt, lane), b, a1);
    }
    int col = nt * 16 + (lane & 15), rb = (lane >> 4) * 8;
    float bv = msg_b2[layer * 256 + col];
    for (int i = 0; i < 8; ++i) {
      atomicAdd(&agg[(size_t)sDst[rb + i] * 256 + col],      a0[i] + bv);
      atomicAdd(&agg[(size_t)sDst[16 + rb + i] * 256 + col], a1[i] + bv);
    }
  }
}

// ---------------------------------------------------------------------------
// Node update: upd = relu([x|agg] @ W1 + b1) @ W2 + b2; x = LN(x + upd)
// ---------------------------------------------------------------------------
__global__ __launch_bounds__(256) void upd_kernel(
    float* __restrict__ x, __bf16* __restrict__ xb, const float* __restrict__ agg,
    const __bf16* __restrict__ updW1f, const float* __restrict__ upd_b1,
    const __bf16* __restrict__ updW2f, const float* __restrict__ upd_b2,
    const float* __restrict__ ln_g, const float* __restrict__ ln_b, int layer) {
  __shared__ __bf16 A[16 * 512];
  __shared__ __bf16 Hd[16 * 256];
  __shared__ float  Y[16 * 256];
  __shared__ float  ps[256], pq[256];
  __shared__ float  sMu[16], sRi[16];
  const int tid = threadIdx.x, lane = tid & 31, wave = tid >> 5;
  const int r0 = blockIdx.x * 16;
  for (int j = 0; j < 16; ++j) {
    int flat = tid + j * 256;
    int r = flat >> 8, c = flat & 255;
    float xv = x[(size_t)(r0 + r) * 256 + c];
    Y[flat]              = xv;
    A[r * 512 + c]       = (__bf16)xv;
    A[r * 512 + 256 + c] = (__bf16)agg[(size_t)(r0 + r) * 256 + c];
  }
  __syncthreads();
  const __bf16* W1 = updW1f + (size_t)layer * (16 * 16 * 512);
  for (int t = 0; t < 2; ++t) {
    int nt = wave * 2 + t;
    v8f acc = {};
    for (int kt = 0; kt < 16; ++kt)
      acc = wmma_bf16(ldsA_frag(A, 512, kt, lane), b_frag(W1, kt, nt, lane), acc);
    int col = nt * 16 + (lane & 15), rb = (lane >> 4) * 8;
    float bv = upd_b1[layer * 256 + col];
    for (int i = 0; i < 8; ++i)
      Hd[(rb + i) * 256 + col] = (__bf16)fmaxf(acc[i] + bv, 0.f);
  }
  __syncthreads();
  const __bf16* W2 = updW2f + (size_t)layer * (8 * 16 * 512);
  for (int t = 0; t < 2; ++t) {
    int nt = wave * 2 + t;
    v8f acc = {};
    for (int kt = 0; kt < 8; ++kt)
      acc = wmma_bf16(ldsA_frag(Hd, 256, kt, lane), b_frag(W2, kt, nt, lane), acc);
    int col = nt * 16 + (lane & 15), rb = (lane >> 4) * 8;
    float bv = upd_b2[layer * 256 + col];
    for (int i = 0; i < 8; ++i)
      Y[(rb + i) * 256 + col] += acc[i] + bv;   // residual
  }
  __syncthreads();
  {
    int r = tid >> 4, seg = tid & 15;
    float s = 0.f, q = 0.f;
    for (int c = seg * 16; c < seg * 16 + 16; ++c) {
      float v = Y[r * 256 + c];
      s += v; q += v * v;
    }
    ps[tid] = s; pq[tid] = q;
  }
  __syncthreads();
  if (tid < 16) {
    float s = 0.f, q = 0.f;
    for (int k = 0; k < 16; ++k) { s += ps[tid * 16 + k]; q += pq[tid * 16 + k]; }
    float mu  = s * (1.f / 256.f);
    float var = q * (1.f / 256.f) - mu * mu;
    sMu[tid] = mu;
    sRi[tid] = rsqrtf(var + 1e-5f);
  }
  __syncthreads();
  {
    int r = tid >> 4, seg = tid & 15;
    float mu = sMu[r], ri = sRi[r];
    for (int c = seg * 16; c < seg * 16 + 16; ++c) {
      float v = (Y[r * 256 + c] - mu) * ri * ln_g[layer * 256 + c] + ln_b[layer * 256 + c];
      size_t idx = (size_t)(r0 + r) * 256 + c;
      x[idx]  = v;
      xb[idx] = (__bf16)v;
    }
  }
}

// ---------------------------------------------------------------------------
// Pooling + readout (tiny, plain f32)
// ---------------------------------------------------------------------------
__global__ __launch_bounds__(256) void pool_scatter(const float* __restrict__ x,
                                                    const long long* __restrict__ batch,
                                                    float* __restrict__ sums,
                                                    float* __restrict__ cnts, int N) {
  int i = blockIdx.x * 256 + threadIdx.x;
  if (i >= N * 256) return;
  int n = i >> 8, c = i & 255;
  int b = (int)batch[n];
  atomicAdd(&sums[b * 256 + c], x[i]);
  if (c == 0) atomicAdd(&cnts[b], 1.0f);
}

__global__ __launch_bounds__(256) void readout1(const float* __restrict__ sums,
                                                const float* __restrict__ cnts,
                                                const float* __restrict__ W1,
                                                const float* __restrict__ b1,
                                                float* __restrict__ hro) {
  __shared__ float prow[256];
  int r = blockIdx.x, c = threadIdx.x;
  prow[c] = sums[r * 256 + c] / cnts[r];
  __syncthreads();
  float acc = b1[c];
  for (int k = 0; k < 256; ++k) acc += prow[k] * W1[k * 256 + c];
  hro[r * 256 + c] = fmaxf(acc, 0.f);
}

__global__ __launch_bounds__(256) void readout2(const float* __restrict__ hro,
                                                const float* __restrict__ W2,
                                                const float* __restrict__ b2,
                                                float* __restrict__ out) {
  int gid = blockIdx.x * 256 + threadIdx.x;
  if (gid >= 64 * 10) return;
  int r = gid / 10, c = gid % 10;
  float acc = b2[c];
  for (int k = 0; k < 256; ++k) acc += hro[r * 256 + k] * W2[k * 10 + c];
  out[gid] = acc;
}

// ---------------------------------------------------------------------------
extern "C" void kernel_launch(void* const* d_in, const int* in_sizes, int n_in,
                              void* d_out, int out_size, void* d_ws, size_t ws_size,
                              hipStream_t stream) {
  const float*     node_features = (const float*)d_in[0];
  const long long* edge_index    = (const long long*)d_in[1];
  const float*     edge_features = (const float*)d_in[2];
  const long long* batch         = (const long long*)d_in[3];
  const float* node_W = (const float*)d_in[4];
  const float* node_b = (const float*)d_in[5];
  const float* edge_W = (const float*)d_in[6];
  const float* edge_b = (const float*)d_in[7];
  const float* msg_W1 = (const float*)d_in[8];
  const float* msg_b1 = (const float*)d_in[9];
  const float* msg_W2 = (const float*)d_in[10];
  const float* msg_b2 = (const float*)d_in[11];
  const float* upd_W1 = (const float*)d_in[12];
  const float* upd_b1 = (const float*)d_in[13];
  const float* upd_W2 = (const float*)d_in[14];
  const float* upd_b2 = (const float*)d_in[15];
  const float* ln_g   = (const float*)d_in[16];
  const float* ln_b   = (const float*)d_in[17];
  const float* ro_W1  = (const float*)d_in[18];
  const float* ro_b1  = (const float*)d_in[19];
  const float* ro_W2  = (const float*)d_in[20];
  const float* ro_b2  = (const float*)d_in[21];

  const int N = in_sizes[0] / 128;   // 10000
  const int E = in_sizes[2] / 64;    // 160000

  // -------- workspace layout (~50 MB) --------
  char*  ws   = (char*)d_ws;
  float* x    = (float*)ws;                                 // N*256 f32
  float* agg  = (float*)(ws + (size_t)N * 256 * 4);         // N*256 f32
  float* sums = (float*)(ws + (size_t)N * 256 * 8);         // 64*256 f32
  float* cnts = sums + 64 * 256;                            // 64 f32
  float* hro  = cnts + 64;                                  // 64*256 f32
  __bf16* wfb    = (__bf16*)(hro + 64 * 256);
  __bf16* nodeWf = wfb;
  __bf16* edgeWf = nodeWf + 128 * 256;
  __bf16* msgW1f = edgeWf + 64 * 256;
  __bf16* msgW2f = msgW1f + (size_t)4 * 768 * 256;
  __bf16* updW1f = msgW2f + (size_t)4 * 256 * 256;
  __bf16* updW2f = updW1f + (size_t)4 * 512 * 256;
  __bf16* xb     = updW2f + (size_t)4 * 256 * 256;          // N*256 bf16
  __bf16* efb    = xb + (size_t)N * 256;                    // E*64 bf16

  // -------- one-time conversions --------
  conv_wfrag<<<(128 * 256 + 255) / 256, 256, 0, stream>>>(node_W, nodeWf, 128, 256);
  conv_wfrag<<<(64 * 256 + 255) / 256, 256, 0, stream>>>(edge_W, edgeWf, 64, 256);
  for (int l = 0; l < 4; ++l) {
    conv_wfrag<<<768, 256, 0, stream>>>(msg_W1 + (size_t)l * 768 * 256,
                                        msgW1f + (size_t)l * 768 * 256, 768, 256);
    conv_wfrag<<<256, 256, 0, stream>>>(msg_W2 + (size_t)l * 256 * 256,
                                        msgW2f + (size_t)l * 256 * 256, 256, 256);
    conv_wfrag<<<512, 256, 0, stream>>>(upd_W1 + (size_t)l * 512 * 256,
                                        updW1f + (size_t)l * 512 * 256, 512, 256);
    conv_wfrag<<<256, 256, 0, stream>>>(upd_W2 + (size_t)l * 256 * 256,
                                        updW2f + (size_t)l * 256 * 256, 256, 256);
  }
  conv_bf16<<<(E * 64 + 255) / 256, 256, 0, stream>>>(edge_features, efb, E * 64);

  // -------- encode + message-passing layers --------
  node_encode<<<N / 16, 256, 0, stream>>>(node_features, nodeWf, node_b, x, xb);
  for (int l = 0; l < 4; ++l) {
    zero_f32<<<(N * 256 + 255) / 256, 256, 0, stream>>>(agg, N * 256);
    msg_kernel<<<E / 32, 256, 0, stream>>>(xb, edge_index, efb, edgeWf, edge_b,
                                           msgW1f, msg_b1, msgW2f, msg_b2,
                                           agg, l, E);
    upd_kernel<<<N / 16, 256, 0, stream>>>(x, xb, agg, updW1f, upd_b1,
                                           updW2f, upd_b2, ln_g, ln_b, l);
  }

  // -------- pooling + readout --------
  zero_f32<<<(64 * 256 + 64 + 255) / 256, 256, 0, stream>>>(sums, 64 * 256 + 64);
  pool_scatter<<<(N * 256 + 255) / 256, 256, 0, stream>>>(x, batch, sums, cnts, N);
  readout1<<<64, 256, 0, stream>>>(sums, cnts, ro_W1, ro_b1, hro);
  readout2<<<3, 256, 0, stream>>>(hro, ro_W2, ro_b2, (float*)d_out);
}
